// TwoLayerLSTM_17892833755221
// MI455X (gfx1250) — matrix operand order repaired and checked by reading, not compiled, not run on hardware
//
#include <hip/hip_runtime.h>
#include <stdint.h>

// ---------------------------------------------------------------------------
// Two-layer LSTM, T=4096 steps, H=2048, D=66.
// Strategy (MI455X / gfx1250, wave32):
//  * Recurrent GEMVs are memory bound: 3 x (8192x2048) weight matrices read
//    every step. bf16 weights = 96 MB/step -> resident in the 192 MB L2.
//  * One persistent kernel runs all 4096 steps with an atomic grid barrier
//    (2 barriers/step); h1/h2 double-buffered in global, staged in LDS.
//  * 16 threads per hidden unit, fp32 accumulation, owner lane does the
//    gate nonlinearities and keeps c1/c2 in registers.
//  * The only real GEMM (out = h2 @ W_lin^T, 4032x66x2048) uses
//    v_wmma_f32_16x16x32_bf16, one 16x16 tile per wave.
// ---------------------------------------------------------------------------

#define H_DIM   2048
#define D_IN    66
#define T_STEPS 4096
#define NBLK    128
#define TPB     256
#define TPU     16                  // threads per hidden unit (NBLK*TPB / H)
#define KITER   (H_DIM / (TPU * 8)) // uint4 loads per row per thread = 16

typedef unsigned short u16_t;
typedef __bf16 v16bf __attribute__((ext_vector_type(16)));
typedef float  v8f   __attribute__((ext_vector_type(8)));

struct alignas(16) U4 { unsigned x, y, z, w; };

// ----------------------------- workspace layout ----------------------------
constexpr size_t SZ_BIGW   = (size_t)4 * H_DIM * H_DIM * sizeof(u16_t); // 32 MB
constexpr size_t OFF_WHH1  = 0;
constexpr size_t OFF_WIH2  = OFF_WHH1 + SZ_BIGW;
constexpr size_t OFF_WHH2  = OFF_WIH2 + SZ_BIGW;
constexpr size_t OFF_WLIN  = OFF_WHH2 + SZ_BIGW;
constexpr size_t SZ_WLIN   = (size_t)D_IN * H_DIM * sizeof(u16_t);      // 264 KB
constexpr size_t OFF_H1    = OFF_WLIN + SZ_WLIN;                        // 2 buffers
constexpr size_t OFF_H2    = OFF_H1 + 2 * H_DIM * sizeof(float);
constexpr size_t OFF_CNT   = OFF_H2 + 2 * H_DIM * sizeof(float);
constexpr size_t OFF_H2HIST= OFF_CNT + 256;                             // T*H bf16

// ------------------------------- helpers -----------------------------------
static __device__ __forceinline__ u16_t f2bf(float f) {      // RNE fp32->bf16
  unsigned u = __float_as_uint(f);
  unsigned r = u + 0x7FFFu + ((u >> 16) & 1u);
  return (u16_t)(r >> 16);
}
static __device__ __forceinline__ float bflo(unsigned u) { return __uint_as_float(u << 16); }
static __device__ __forceinline__ float bfhi(unsigned u) { return __uint_as_float(u & 0xFFFF0000u); }
static __device__ __forceinline__ float sigm(float x) { return 1.0f / (1.0f + __expf(-x)); }

// bf16 row (length 2048) dot fp32 LDS vector, this thread's 1/16 slice.
static __device__ __forceinline__ float dot_row_bf16(const u16_t* __restrict__ row,
                                                     const float* __restrict__ hs,
                                                     int s) {
  float acc = 0.0f;
#pragma unroll
  for (int i = 0; i < KITER; ++i) {
    const int k = i * (TPU * 8) + s * 8;                 // 128-elem stripes
    const uint4  w  = *(const uint4 *)(row + k);         // 8 bf16, b128 load
    const float4 ha = *(const float4*)(hs + k);
    const float4 hb = *(const float4*)(hs + k + 4);
    acc = fmaf(bflo(w.x), ha.x, acc); acc = fmaf(bfhi(w.x), ha.y, acc);
    acc = fmaf(bflo(w.y), ha.z, acc); acc = fmaf(bfhi(w.y), ha.w, acc);
    acc = fmaf(bflo(w.z), hb.x, acc); acc = fmaf(bfhi(w.z), hb.y, acc);
    acc = fmaf(bflo(w.w), hb.z, acc); acc = fmaf(bfhi(w.w), hb.w, acc);
  }
  return acc;
}

static __device__ __forceinline__ void grid_sync(unsigned* cnt, unsigned target) {
  __syncthreads();
  if (threadIdx.x == 0) {
    __threadfence();                       // release stores
    atomicAdd(cnt, 1u);
    while (__hip_atomic_load(cnt, __ATOMIC_ACQUIRE, __HIP_MEMORY_SCOPE_AGENT) < target)
      __builtin_amdgcn_s_sleep(2);
  }
  __syncthreads();
  __threadfence();                         // acquire for every wave
}

// ------------------------- kernel 1: weight prep ---------------------------
__global__ void lstm_prep_kernel(const float* __restrict__ Whh1,
                                 const float* __restrict__ Wih2,
                                 const float* __restrict__ Whh2,
                                 const float* __restrict__ Wlin,
                                 u16_t* __restrict__ bWhh1,
                                 u16_t* __restrict__ bWih2,
                                 u16_t* __restrict__ bWhh2,
                                 u16_t* __restrict__ bWlin,
                                 unsigned* __restrict__ cnt) {
  if (blockIdx.x == 0 && threadIdx.x == 0) *cnt = 0u;   // reset grid barrier
  const size_t n_big  = (size_t)4 * H_DIM * H_DIM;
  const size_t n_lin  = (size_t)D_IN * H_DIM;
  const size_t stride = (size_t)gridDim.x * blockDim.x;
  for (size_t i = (size_t)blockIdx.x * blockDim.x + threadIdx.x; i < n_big; i += stride) {
    bWhh1[i] = f2bf(Whh1[i]);
    bWih2[i] = f2bf(Wih2[i]);
    bWhh2[i] = f2bf(Whh2[i]);
    if (i < n_lin) bWlin[i] = f2bf(Wlin[i]);
  }
}

// -------------------- kernel 2: persistent recurrence ----------------------
__global__ __launch_bounds__(TPB) void lstm_seq_kernel(
    const float* __restrict__ y,
    const float* __restrict__ Wih1,
    const float* __restrict__ bih1, const float* __restrict__ bhh1,
    const float* __restrict__ bih2, const float* __restrict__ bhh2,
    const u16_t* __restrict__ bWhh1,
    const u16_t* __restrict__ bWih2,
    const u16_t* __restrict__ bWhh2,
    float* __restrict__ h1g,            // 2 x H double buffer
    float* __restrict__ h2g,            // 2 x H double buffer
    u16_t* __restrict__ h2hist,         // T x H bf16
    unsigned* __restrict__ cnt) {
  __shared__ float sh1[H_DIM];
  __shared__ float sh2[H_DIM];
  __shared__ float sx[D_IN];

  const int tid  = threadIdx.x;
  const int gtid = blockIdx.x * TPB + tid;
  const int j    = gtid >> 4;           // hidden unit 0..2047
  const int s    = gtid & (TPU - 1);    // slice within unit

  float c1 = 0.0f, c2 = 0.0f;
  unsigned bar = 0;

  if (s == 0) {                          // zero both h double-buffers
    h1g[j] = 0.0f; h1g[H_DIM + j] = 0.0f;
    h2g[j] = 0.0f; h2g[H_DIM + j] = 0.0f;
  }
  bar += 1; grid_sync(cnt, bar * NBLK);

  for (int t = 0; t < T_STEPS; ++t) {
    const int cur = t & 1, prv = cur ^ 1;

    // ---------------- layer 1: gates = x W_ih1^T + h1 W_hh1^T --------------
    for (int k = tid; k < H_DIM; k += TPB) sh1[k] = h1g[prv * H_DIM + k];
    for (int k = tid; k < D_IN;  k += TPB) sx[k]  = y[(size_t)t * D_IN + k];
    __syncthreads();

    float acc[4];
#pragma unroll
    for (int g = 0; g < 4; ++g) {
      float a = dot_row_bf16(bWhh1 + (size_t)(g * H_DIM + j) * H_DIM, sh1, s);
      const float* wx = Wih1 + (size_t)(g * H_DIM + j) * D_IN;   // fp32, tiny
      for (int k = s; k < D_IN; k += TPU) a = fmaf(sx[k], wx[k], a);
      acc[g] = a;
    }
#pragma unroll
    for (int g = 0; g < 4; ++g)
#pragma unroll
      for (int m = TPU / 2; m >= 1; m >>= 1)
        acc[g] += __shfl_xor(acc[g], m, 32);

    if (s == 0) {
      float iv = sigm (acc[0] + bih1[j]              + bhh1[j]);
      float fv = sigm (acc[1] + bih1[H_DIM + j]      + bhh1[H_DIM + j]);
      float gv = tanhf(acc[2] + bih1[2 * H_DIM + j]  + bhh1[2 * H_DIM + j]);
      float ov = sigm (acc[3] + bih1[3 * H_DIM + j]  + bhh1[3 * H_DIM + j]);
      c1 = fv * c1 + iv * gv;
      h1g[cur * H_DIM + j] = ov * tanhf(c1);
    }
    bar += 1; grid_sync(cnt, bar * NBLK);

    // ---------------- layer 2: gates = h1 W_ih2^T + h2 W_hh2^T -------------
    for (int k = tid; k < H_DIM; k += TPB) {
      sh1[k] = h1g[cur * H_DIM + k];
      sh2[k] = h2g[prv * H_DIM + k];
    }
    __syncthreads();

#pragma unroll
    for (int g = 0; g < 4; ++g)
      acc[g] = dot_row_bf16(bWih2 + (size_t)(g * H_DIM + j) * H_DIM, sh1, s)
             + dot_row_bf16(bWhh2 + (size_t)(g * H_DIM + j) * H_DIM, sh2, s);
#pragma unroll
    for (int g = 0; g < 4; ++g)
#pragma unroll
      for (int m = TPU / 2; m >= 1; m >>= 1)
        acc[g] += __shfl_xor(acc[g], m, 32);

    if (s == 0) {
      float iv = sigm (acc[0] + bih2[j]              + bhh2[j]);
      float fv = sigm (acc[1] + bih2[H_DIM + j]      + bhh2[H_DIM + j]);
      float gv = tanhf(acc[2] + bih2[2 * H_DIM + j]  + bhh2[2 * H_DIM + j]);
      float ov = sigm (acc[3] + bih2[3 * H_DIM + j]  + bhh2[3 * H_DIM + j]);
      c2 = fv * c2 + iv * gv;
      float h2v = ov * tanhf(c2);
      h2g[cur * H_DIM + j] = h2v;
      h2hist[(size_t)t * H_DIM + j] = f2bf(h2v);
    }
    bar += 1; grid_sync(cnt, bar * NBLK);
  }
}

// --------------- kernel 3: WMMA projection out = H2 W_lin^T + b ------------
__global__ __launch_bounds__(256) void lstm_proj_kernel(
    const u16_t* __restrict__ h2hist,   // T x H bf16 (row major)
    const u16_t* __restrict__ bWlin,    // D_IN x H bf16 (row major)
    const float* __restrict__ blin,
    float* __restrict__ out,            // rows_out x D_IN fp32
    int rows_out, int pre_len) {
  const int lane   = threadIdx.x & 31;
  const int wv     = threadIdx.x >> 5;
  const int mtiles = (rows_out + 15) >> 4;
  const int ntiles = (D_IN + 15) >> 4;      // 5
  const int tile   = blockIdx.x * 8 + wv;   // one 16x16 tile per wave
  if (tile >= mtiles * ntiles) return;      // wave-uniform exit
  const int mt = tile / ntiles;
  const int nt = tile % ntiles;

  const int half = lane >> 4;               // 0: lanes 0-15, 1: lanes 16-31
  const int l15  = lane & 15;

  // A fragment source: H2 rows (clamped so all lanes stay in bounds)
  int arow = pre_len + mt * 16 + l15;
  if (arow > T_STEPS - 1) arow = T_STEPS - 1;
  const u16_t* aptr = h2hist + (size_t)arow * H_DIM;

  // B fragment source: W_lin row = output column (clamped; masked at store)
  int bcol = nt * 16 + l15;
  if (bcol > D_IN - 1) bcol = D_IN - 1;
  const u16_t* bptr = bWlin + (size_t)bcol * H_DIM;

  union Frag { v16bf v; U4 q[2]; } A, B;
  v8f acc = {};

  for (int kt = 0; kt < H_DIM / 32; ++kt) {
    const int kb = kt * 32;
    // A (16x32 bf16): lanes0-15 K={0..7,16..23}, lanes16-31 K={8..15,24..31}
    A.q[0] = *(const U4*)(aptr + kb + half * 8);
    A.q[1] = *(const U4*)(aptr + kb + half * 8 + 16);
    // B (32x16 bf16): lane = column; lanes0-15 K={0..7,16..23}, lanes16-31 K={8..15,24..31}
    B.q[0] = *(const U4*)(bptr + kb + half * 8);
    B.q[1] = *(const U4*)(bptr + kb + half * 8 + 16);
    acc = __builtin_amdgcn_wmma_f32_16x16x32_bf16(false, A.v, false, B.v,
                                                  (short)0, acc, false, false);
  }

  // C/D layout: VGPR r -> M = r + half*8, N = lane&15
  const int col = nt * 16 + l15;
  if (col < D_IN) {
    const float bb = blin[col];
#pragma unroll
    for (int r = 0; r < 8; ++r) {
      const int row = mt * 16 + r + half * 8;
      if (row < rows_out) out[(size_t)row * D_IN + col] = acc[r] + bb;
    }
  }
}

// ------------------------------- launcher ----------------------------------
extern "C" void kernel_launch(void* const* d_in, const int* in_sizes, int n_in,
                              void* d_out, int out_size, void* d_ws, size_t ws_size,
                              hipStream_t stream) {
  const float* y    = (const float*)d_in[0];
  const float* Wih1 = (const float*)d_in[1];
  const float* Whh1 = (const float*)d_in[2];
  const float* bih1 = (const float*)d_in[3];
  const float* bhh1 = (const float*)d_in[4];
  const float* Wih2 = (const float*)d_in[5];
  const float* Whh2 = (const float*)d_in[6];
  const float* bih2 = (const float*)d_in[7];
  const float* bhh2 = (const float*)d_in[8];
  const float* Wlin = (const float*)d_in[9];
  const float* blin = (const float*)d_in[10];
  float* out = (float*)d_out;

  char* ws = (char*)d_ws;
  u16_t*    bWhh1  = (u16_t*)(ws + OFF_WHH1);
  u16_t*    bWih2  = (u16_t*)(ws + OFF_WIH2);
  u16_t*    bWhh2  = (u16_t*)(ws + OFF_WHH2);
  u16_t*    bWlin  = (u16_t*)(ws + OFF_WLIN);
  float*    h1g    = (float*)(ws + OFF_H1);
  float*    h2g    = (float*)(ws + OFF_H2);
  unsigned* cnt    = (unsigned*)(ws + OFF_CNT);
  u16_t*    h2hist = (u16_t*)(ws + OFF_H2HIST);

  const int rows_out = out_size / D_IN;         // 4032
  const int pre_len  = T_STEPS - rows_out;      // 64

  lstm_prep_kernel<<<1024, 256, 0, stream>>>(Whh1, Wih2, Whh2, Wlin,
                                             bWhh1, bWih2, bWhh2, bWlin, cnt);

  lstm_seq_kernel<<<NBLK, TPB, 0, stream>>>(y, Wih1, bih1, bhh1, bih2, bhh2,
                                            bWhh1, bWih2, bWhh2,
                                            h1g, h2g, h2hist, cnt);

  const int mtiles = (rows_out + 15) / 16;
  const int ntiles = (D_IN + 15) / 16;
  const int tiles  = mtiles * ntiles;
  lstm_proj_kernel<<<(tiles + 7) / 8, 256, 0, stream>>>(h2hist, bWlin, blin,
                                                        out, rows_out, pre_len);
  (void)in_sizes; (void)n_in; (void)ws_size;
}
